// Attention_42374147342446
// MI455X (gfx1250) — matrix operand order repaired and checked by reading, not compiled, not run on hardware
//
#include <hip/hip_runtime.h>
#include <stdint.h>

// Problem constants (match reference)
#define Bsz 4
#define Nn  1024
#define Ff  512
#define Hh  8
#define Dd  64
#define Ee  16
#define HD  512          // H*D == F
#define Mrows (Bsz*Nn)   // 4096

typedef __attribute__((ext_vector_type(16))) _Float16 v16h;
typedef __attribute__((ext_vector_type(8)))  _Float16 v8h;
typedef __attribute__((ext_vector_type(8)))  float    v8f;

__device__ __forceinline__ v8f wmma_f16(v16h a, v16h b, v8f c) {
  // D = A(16x32 f16) x B(32x16 f16) + C(16x16 f32)
  return __builtin_amdgcn_wmma_f32_16x16x32_f16(
      /*neg_a=*/false, a, /*neg_b=*/false, b,
      /*c_mod=*/(short)0, c, /*reuse_a=*/false, /*reuse_b=*/false);
}

// ---- fragment loaders (contiguous k order k = 16*half + s, same in A and B, wave32) ----

// 16-bit A matrix 16x32 (MxK), row-major source with `stride` halves per row.
// lanes 0-15: row=lane, halves K+0..15; lanes 16-31: row=lane-16,
// halves K+16..31. Contiguous 32B per lane.
__device__ __forceinline__ v16h load_a_frag(const _Float16* base, int stride,
                                            int kbase, int lane) {
  const int half = lane >> 4;
  const int row  = lane & 15;
  const _Float16* p = base + (size_t)row * stride + kbase + half * 16;
  v8h lo = *(const v8h*)(p);
  v8h hi = *(const v8h*)(p + 8);
  v16h r;
#pragma unroll
  for (int i = 0; i < 8; ++i) { r[i] = lo[i]; r[8 + i] = hi[i]; }
  return r;
}

// 16-bit B matrix 32x16 (KxN) where memory holds, per logical column n, the
// K values contiguously (i.e. B^T row-major, stride = col_stride halves).
// lanes 0-15 hold K=kbase+0..15 of col n=nbase+lane; lanes 16-31 K=+16..31.
// Contiguous 32B per lane.
__device__ __forceinline__ v16h load_b_frag_tr(const _Float16* base,
                                               int col_stride, int kbase,
                                               int nbase, int lane) {
  const int half = lane >> 4;
  const int col  = (lane & 15) + nbase;
  const _Float16* p = base + (size_t)col * col_stride + kbase + half * 16;
  v8h lo = *(const v8h*)(p);
  v8h hi = *(const v8h*)(p + 8);
  v16h r;
#pragma unroll
  for (int i = 0; i < 8; ++i) { r[i] = lo[i]; r[8 + i] = hi[i]; }
  return r;
}

// low 32 bits of a generic pointer to __shared__ == wave-relative LDS address
__device__ __forceinline__ uint32_t lds_addr32(const void* p) {
  return (uint32_t)(uintptr_t)p;
}

// ---- kernel 1: fp32 -> f16 TRANSPOSED convert (weights: W[K,N] -> WT[N,K]) ----
__global__ __launch_bounds__(256) void cvt_transpose_kernel(
    const float* __restrict__ s, _Float16* __restrict__ dT) {
  __shared__ _Float16 tile[64][65];
  const int bx = blockIdx.x & 7;   // 64-col tile
  const int by = blockIdx.x >> 3;  // 64-row tile
  const int r0 = by * 64, c0 = bx * 64;
  for (int t = threadIdx.x; t < 4096; t += 256) {
    int r = t >> 6, c = t & 63;  // coalesced read along c
    tile[r][c] = (_Float16)s[(size_t)(r0 + r) * HD + c0 + c];
  }
  __syncthreads();
  for (int t = threadIdx.x; t < 4096; t += 256) {
    int c = t >> 6, r = t & 63;  // coalesced write along r
    dT[(size_t)(c0 + c) * HD + r0 + r] = tile[r][c];
  }
}

// ---- kernel 2: LayerNorm over F=512, write f16 ----
__global__ __launch_bounds__(256) void ln_kernel(const float* __restrict__ x,
                                                 const float* __restrict__ gam,
                                                 const float* __restrict__ bet,
                                                 _Float16* __restrict__ out) {
  __shared__ float rs[256], rs2[256];
  const int row = blockIdx.x;
  const int tid = threadIdx.x;
  const float* xr = x + (size_t)row * Ff;
  float v0 = xr[tid], v1 = xr[tid + 256];
  rs[tid]  = v0 + v1;
  rs2[tid] = v0 * v0 + v1 * v1;
  __syncthreads();
  for (int off = 128; off > 0; off >>= 1) {
    if (tid < off) { rs[tid] += rs[tid + off]; rs2[tid] += rs2[tid + off]; }
    __syncthreads();
  }
  float mu  = rs[0] * (1.0f / Ff);
  float var = rs2[0] * (1.0f / Ff) - mu * mu;
  float inv = rsqrtf(var + 1e-5f);
  out[(size_t)row * Ff + tid]       = (_Float16)((v0 - mu) * inv * gam[tid] + bet[tid]);
  out[(size_t)row * Ff + tid + 256] = (_Float16)((v1 - mu) * inv * gam[tid + 256] + bet[tid + 256]);
}

// ---- kernel 3/5: WMMA GEMM, M=4096 K=512 N=512, weights pre-transposed.
// No LDS: A fragments and BT fragments are contiguous 16/32B-per-lane loads.
// out16 + !vtrans : f16 row-major output (q/k projections)
// out16 + vtrans  : f16 output transposed per head -> [B,H,D,N] (v projection)
// out16 == null   : f32 output + residual add (final projection)
__global__ __launch_bounds__(256) void gemm512_kernel(
    const _Float16* __restrict__ A, const _Float16* __restrict__ BT,
    _Float16* __restrict__ out16, float* __restrict__ out32,
    const float* __restrict__ residual, int vtrans) {
  const int tid  = threadIdx.x;
  const int wave = tid >> 5, lane = tid & 31;
  const int mbase = blockIdx.x * 128 + wave * 16;
  const int nb    = blockIdx.y * 64;
  v8f acc[4] = {{}, {}, {}, {}};

  const _Float16* arow = A + (size_t)mbase * Ff;
  for (int kk = 0; kk < Ff; kk += 32) {
    v16h a = load_a_frag(arow, Ff, kk, lane);
#pragma unroll
    for (int t = 0; t < 4; ++t) {
      v16h b = load_b_frag_tr(BT, HD, kk, nb + t * 16, lane);
      acc[t] = wmma_f16(a, b, acc[t]);
    }
  }

  const int rowoff = (lane >> 4) * 8;
  const int col    = lane & 15;
  if (out16 && vtrans) {
    // store V^T: [B,H,D,N]; 8 accumulator rows are contiguous in i -> v8h pack
    const int mg = mbase + rowoff;
    const int bidx = mg / Nn, i0 = mg - bidx * Nn;
#pragma unroll
    for (int t = 0; t < 4; ++t) {
      const int cidx = nb + t * 16 + col;  // = h*64 + d
      const int h = cidx >> 6, d = cidx & 63;
      v8h pk;
#pragma unroll
      for (int r = 0; r < 8; ++r) pk[r] = (_Float16)acc[t][r];
      *(v8h*)(out16 + (((size_t)bidx * Hh + h) * Dd + d) * Nn + i0) = pk;
    }
  } else {
#pragma unroll
    for (int r = 0; r < 8; ++r) {
      const int row = mbase + rowoff + r;
      const size_t o = (size_t)row * HD + nb;
      if (out16) {
#pragma unroll
        for (int t = 0; t < 4; ++t)
          out16[o + t * 16 + col] = (_Float16)acc[t][r];
      } else {
#pragma unroll
        for (int t = 0; t < 4; ++t)
          out32[o + t * 16 + col] = acc[t][r] + residual[o + t * 16 + col];
      }
    }
  }
}

// ---- kernel 4: fused edge-bias attention with online softmax ----
// Block = (b, 16-row i-tile). 8 waves = 8 heads share the edge/mask tiles.
// Edge tile streamed with CDNA5 async global->LDS copies (ASYNCcnt).
__global__ __launch_bounds__(256) void attn_kernel(
    const _Float16* __restrict__ q, const _Float16* __restrict__ kmat,
    const _Float16* __restrict__ vT, const float* __restrict__ edge,
    const float* __restrict__ mask, const float* __restrict__ We,
    _Float16* __restrict__ attn16) {
  __shared__ __align__(16) float lds_edge[16 * 32 * Ee];  // 32 KB
  __shared__ float    lds_mask[16 * 32];                  //  2 KB
  __shared__ float    lds_bias[16 * 32 * Hh];             // 16 KB
  __shared__ float    lds_we[Ee * Hh];                    // 0.5 KB
  __shared__ __align__(16) _Float16 lds_p[8][16 * 32];    //  8 KB

  const int tid  = threadIdx.x;
  const int wave = tid >> 5, lane = tid & 31;
  const int h     = wave;
  const int b     = blockIdx.x >> 6;        // N/16 = 64 i-tiles per batch
  const int ibase = (blockIdx.x & 63) * 16;

  if (tid < Ee * Hh) lds_we[tid] = We[tid];

  const _Float16* qb = q + ((size_t)(b * Nn + ibase)) * HD + h * Dd;
  v16h qa0 = load_a_frag(qb, HD, 0, lane);
  v16h qa1 = load_a_frag(qb, HD, 32, lane);

  v8f acc[4] = {{}, {}, {}, {}};
  float mrow[8], lrow[8];
#pragma unroll
  for (int r = 0; r < 8; ++r) { mrow[r] = -3.0e38f; lrow[r] = 0.f; }

  const int rowoff = (lane >> 4) * 8;
  const int col    = lane & 15;
  const uint32_t lds_edge_base = lds_addr32(&lds_edge[0]);

  for (int jt = 0; jt < Nn / 32; ++jt) {
    const int jbase = jt * 32;
    const float* esrc = edge + (((size_t)(b * Nn + ibase)) * Nn + jbase) * Ee;
    __syncthreads();
    // edge tile: 2048 x 16B lane-copies, global -> LDS, no VGPR round trip
    for (int t = tid; t < 16 * 32 * Ee / 4; t += 256) {
      int i   = t / (32 * Ee / 4);
      int rem = t - i * (32 * Ee / 4);
      uint32_t loff = lds_edge_base + (uint32_t)t * 16u;
      unsigned long long ga =
          (unsigned long long)(uintptr_t)(esrc + (size_t)i * Nn * Ee + rem * 4);
      asm volatile("global_load_async_to_lds_b128 %0, %1, off"
                   :: "v"(loff), "v"(ga) : "memory");
    }
    const float* msrc = mask + ((size_t)b * Nn + ibase) * Nn + jbase;
    for (int t = tid; t < 16 * 32; t += 256) {
      int i = t >> 5, j = t & 31;
      lds_mask[t] = msrc[(size_t)i * Nn + j];
    }
    if (jt + 1 < Nn / 32)  // global_prefetch_b8: pull next edge tile into GL2
      __builtin_prefetch(esrc + 32 * Ee + (size_t)(tid & 15) * Nn * Ee +
                             (tid >> 4) * 32, 0, 1);
    asm volatile("s_wait_asynccnt 0x0" ::: "memory");
    __syncthreads();

    // bias[i,j,h] = edge[i,j,:] . We[:,h] — each (i,j) computed once for all 8 h
    for (int t = tid; t < 16 * 32; t += 256) {
      const float* ef = &lds_edge[t * Ee];
      float bh[Hh];
#pragma unroll
      for (int hh = 0; hh < Hh; ++hh) bh[hh] = 0.f;
#pragma unroll
      for (int e = 0; e < Ee; ++e) {
        float ev = ef[e];
#pragma unroll
        for (int hh = 0; hh < Hh; ++hh) bh[hh] += ev * lds_we[e * Hh + hh];
      }
#pragma unroll
      for (int hh = 0; hh < Hh; ++hh) lds_bias[t * Hh + hh] = bh[hh];
    }
    __syncthreads();

    // S = Q K^T  (K row-major in D => contiguous B^T fragment loads)
    const _Float16* kb = kmat + ((size_t)(b * Nn + jbase)) * HD + h * Dd;
    v8f s0 = {}, s1 = {};
    s0 = wmma_f16(qa0, load_b_frag_tr(kb, HD, 0, 0, lane), s0);
    s0 = wmma_f16(qa1, load_b_frag_tr(kb, HD, 32, 0, lane), s0);
    s1 = wmma_f16(qa0, load_b_frag_tr(kb, HD, 0, 16, lane), s1);
    s1 = wmma_f16(qa1, load_b_frag_tr(kb, HD, 32, 16, lane), s1);

    // online softmax over j; denominator is UNmasked, mask applied to P only
#pragma unroll
    for (int r = 0; r < 8; ++r) {
      const int row = rowoff + r;  // C layout: lanes<16 rows 0..7, >=16 rows 8..15
      float x0 = s0[r] + lds_bias[(row * 32 + col) * Hh + h];
      float x1 = s1[r] + lds_bias[(row * 32 + col + 16) * Hh + h];
      float tmax = fmaxf(x0, x1);
      tmax = fmaxf(tmax, __shfl_xor(tmax, 1, 32));
      tmax = fmaxf(tmax, __shfl_xor(tmax, 2, 32));
      tmax = fmaxf(tmax, __shfl_xor(tmax, 4, 32));
      tmax = fmaxf(tmax, __shfl_xor(tmax, 8, 32));
      float nm   = fmaxf(mrow[r], tmax);
      float corr = __expf(mrow[r] - nm);
      float p0   = __expf(x0 - nm);
      float p1   = __expf(x1 - nm);
      float ps = p0 + p1;
      ps += __shfl_xor(ps, 1, 32);
      ps += __shfl_xor(ps, 2, 32);
      ps += __shfl_xor(ps, 4, 32);
      ps += __shfl_xor(ps, 8, 32);
      lrow[r] = lrow[r] * corr + ps;
      mrow[r] = nm;
#pragma unroll
      for (int t = 0; t < 4; ++t) acc[t][r] *= corr;
      lds_p[wave][row * 32 + col]      = (_Float16)(p0 * lds_mask[row * 32 + col]);
      lds_p[wave][row * 32 + col + 16] = (_Float16)(p1 * lds_mask[row * 32 + col + 16]);
    }
    // re-shape masked P (C layout) -> A fragment via per-wave LDS tile
    v16h pa = load_a_frag(&lds_p[wave][0], 32, 0, lane);
    // V^T layout [B,H,D,N]: j contiguous -> contiguous B fragments
    const _Float16* vb = vT + ((size_t)(b * Hh + h) * Dd) * Nn;
#pragma unroll
    for (int t = 0; t < 4; ++t)
      acc[t] = wmma_f16(pa, load_b_frag_tr(vb, Nn, jbase, t * 16, lane), acc[t]);
  }

  // epilogue: /l * (1/sqrt(D)); store f16 [M,512] at head slice
#pragma unroll
  for (int r = 0; r < 8; ++r) {
    const int row = ibase + rowoff + r;
    const float inv = 0.125f / lrow[r];
    const size_t o = ((size_t)(b * Nn) + row) * HD + h * Dd;
#pragma unroll
    for (int t = 0; t < 4; ++t)
      attn16[o + t * 16 + col] = (_Float16)(acc[t][r] * inv);
  }
}

extern "C" void kernel_launch(void* const* d_in, const int* in_sizes, int n_in,
                              void* d_out, int out_size, void* d_ws,
                              size_t ws_size, hipStream_t stream) {
  (void)in_sizes; (void)n_in; (void)out_size; (void)ws_size;
  const float* x    = (const float*)d_in[0];
  const float* edge = (const float*)d_in[1];
  const float* mask = (const float*)d_in[2];
  const float* gam  = (const float*)d_in[3];
  const float* bet  = (const float*)d_in[4];
  const float* Wq   = (const float*)d_in[5];
  const float* Wk   = (const float*)d_in[6];
  const float* Wv   = (const float*)d_in[7];
  const float* We   = (const float*)d_in[8];
  const float* Wo   = (const float*)d_in[9];
  float* out = (float*)d_out;

  // workspace layout (~22 MB, all 256B-aligned)
  char* ws = (char*)d_ws;
  _Float16* r16   = (_Float16*)(ws);                      // 4 MB: LN(x) f16
  _Float16* wq16t = (_Float16*)(ws + (size_t)(4 << 20));  // 4x 512KB W^T f16
  _Float16* wk16t = wq16t + 512 * 512;
  _Float16* wv16t = wk16t + 512 * 512;
  _Float16* wo16t = wv16t + 512 * 512;
  _Float16* q16   = wo16t + 512 * 512;                    // 4 MB each
  _Float16* k16   = q16 + (size_t)Mrows * HD;
  _Float16* v16t  = k16 + (size_t)Mrows * HD;             // [B,H,D,N]
  _Float16* a16   = v16t + (size_t)Mrows * HD;

  cvt_transpose_kernel<<<64, 256, 0, stream>>>(Wq, wq16t);
  cvt_transpose_kernel<<<64, 256, 0, stream>>>(Wk, wk16t);
  cvt_transpose_kernel<<<64, 256, 0, stream>>>(Wv, wv16t);
  cvt_transpose_kernel<<<64, 256, 0, stream>>>(Wo, wo16t);

  ln_kernel<<<Mrows, 256, 0, stream>>>(x, gam, bet, r16);

  dim3 ggrid(Mrows / 128, HD / 64);  // 32 x 8
  gemm512_kernel<<<ggrid, 256, 0, stream>>>(r16, wq16t, q16, nullptr, nullptr, 0);
  gemm512_kernel<<<ggrid, 256, 0, stream>>>(r16, wk16t, k16, nullptr, nullptr, 0);
  gemm512_kernel<<<ggrid, 256, 0, stream>>>(r16, wv16t, v16t, nullptr, nullptr, 1);

  attn_kernel<<<Bsz * (Nn / 16), 256, 0, stream>>>(q16, k16, v16t, edge, mask,
                                                   We, a16);

  gemm512_kernel<<<ggrid, 256, 0, stream>>>(a16, wo16t, nullptr, out, x, 0);
}
